// PointNet2InstanceSegmentation_20160576487545
// MI455X (gfx1250) — matrix operand order repaired and checked by reading, not compiled, not run on hardware
//
#include <hip/hip_runtime.h>

// ============================================================================
// PointNet++ instance-seg forward, MI455X (gfx1250, wave32, WMMA).
//
// Live dataflow (dead code removed: FP modules 0-2 and SA modules 1-3 do not
// influence the reference outputs):
//   points (4,8192,3)
//     -> FPS 2048 -> xyz1, ball query r=0.1 ns=32 -> grouped rel-xyz (R1,32
//        zero-padded so K%32==0)
//     -> SA1 MLP [3->32->32->64] (WMMA GEMM, fused BN+ReLU) -> maxpool -> f1
//     -> 3-NN inverse-distance interp (2048 -> 8192) -> X (RF,64)
//     -> FP MLP [64->128->128] -> feats (RF,128)
//     -> 3 heads: [128->256->128] BN+ReLU, then linear to 13 / 32 / 3
//     -> emb L2-normalized; d_out = [sem | emb | ctr] flat.
//
// GEMM data path (guard-free K loop):
//   A: clause of MT unguarded global_load_b128 -> cvt_pk_bf16 -> ds_store_b64;
//      fragments read as 2x ds_load_b128 (contiguous k order, same as B repack).
//   B: weights pre-repacked to bf16 per-lane fragment order in workspace ->
//      2x global_load_b128 per k-tile, no LDS.
//   MT=4 M-tiles per block => 4 v_wmma per barrier per wave.
// ============================================================================

typedef __attribute__((ext_vector_type(16))) __bf16 v16bf;
typedef __attribute__((ext_vector_type(8)))  __bf16 v8bf;
typedef __attribute__((ext_vector_type(4)))  __bf16 v4bf;
typedef __attribute__((ext_vector_type(8)))  float  v8f;

#define WAVES_PER_BLOCK 4
#define MT 4                 // M-tiles (16 rows each) per block

// ---------------------------------------------------------------------------
// Fold BatchNorm (eval) into per-channel scale/bias.
// ---------------------------------------------------------------------------
__global__ void fold_bn_kernel(const float* __restrict__ b, const float* __restrict__ beta,
                               const float* __restrict__ gamma, const float* __restrict__ mean,
                               const float* __restrict__ var, float* __restrict__ scale,
                               float* __restrict__ bias, int C)
{
    int c = blockIdx.x * blockDim.x + threadIdx.x;
    if (c >= C) return;
    float s = gamma[c] / sqrtf(var[c] + 1e-5f);
    scale[c] = s;
    bias[c]  = (b[c] - mean[c]) * s + beta[c];
}

// ---------------------------------------------------------------------------
// Repack an f32 weight matrix (K,N) into bf16 WMMA B-fragments, zero-padded
// to whole 32x16 tiles. Element ((nt*kTiles+kt)*32+lane)*16+e holds
// W[kt*32 + (lane>>4)*16 + e, nt*16 + (lane&15)]  (contiguous k order, matches A reads).
// ---------------------------------------------------------------------------
__global__ void repack_w_bf16_kernel(const float* __restrict__ W, int K, int N,
                                     int kTiles, int nTiles, __bf16* __restrict__ out)
{
    int gid = blockIdx.x * blockDim.x + threadIdx.x;
    int total = nTiles * kTiles * 512;
    if (gid >= total) return;
    int e    = gid & 15;
    int lane = (gid >> 4) & 31;
    int kt   = (gid >> 9) % kTiles;
    int nt   = (gid >> 9) / kTiles;
    int k = kt * 32 + (lane >> 4) * 16 + e;
    int n = nt * 16 + (lane & 15);
    float v = (k < K && n < N) ? W[(size_t)k * N + n] : 0.0f;
    out[gid] = (__bf16)v;
}

// ---------------------------------------------------------------------------
// Farthest point sampling, deterministic start at index 0, first-max ties.
// ---------------------------------------------------------------------------
__global__ __launch_bounds__(1024) void fps_kernel(const float* __restrict__ xyz,
                                                   int N, int npoint,
                                                   int* __restrict__ outIdx,
                                                   float* __restrict__ distWs)
{
    int b = blockIdx.x;
    const float* p = xyz + (size_t)b * N * 3;
    float* dist = distWs + (size_t)b * N;
    int tid = threadIdx.x;

    __shared__ float s_val[32];
    __shared__ int   s_idx[32];
    __shared__ int   s_far;

    for (int i = tid; i < N; i += 1024) dist[i] = 1e10f;
    if (tid == 0) s_far = 0;
    __syncthreads();

    for (int j = 0; j < npoint; ++j) {
        int far = s_far;
        if (tid == 0) outIdx[(size_t)b * npoint + j] = far;
        float fx = p[3 * far + 0], fy = p[3 * far + 1], fz = p[3 * far + 2];

        float best  = -1.0f;
        int   bestI = 0x7fffffff;
        for (int i = tid; i < N; i += 1024) {
            float dx = p[3 * i + 0] - fx;
            float dy = p[3 * i + 1] - fy;
            float dz = p[3 * i + 2] - fz;
            float dd = dx * dx + dy * dy + dz * dz;
            float nw = fminf(dist[i], dd);
            dist[i] = nw;
            if (nw > best || (nw == best && i < bestI)) { best = nw; bestI = i; }
        }
        for (int off = 16; off > 0; off >>= 1) {
            float ov = __shfl_xor(best, off, 32);
            int   oi = __shfl_xor(bestI, off, 32);
            if (ov > best || (ov == best && oi < bestI)) { best = ov; bestI = oi; }
        }
        if ((tid & 31) == 0) { s_val[tid >> 5] = best; s_idx[tid >> 5] = bestI; }
        __syncthreads();
        if (tid < 32) {
            best = s_val[tid]; bestI = s_idx[tid];
            for (int off = 16; off > 0; off >>= 1) {
                float ov = __shfl_xor(best, off, 32);
                int   oi = __shfl_xor(bestI, off, 32);
                if (ov > best || (ov == best && oi < bestI)) { best = ov; bestI = oi; }
            }
            if (tid == 0) s_far = bestI;
        }
        __syncthreads();
    }
}

// ---------------------------------------------------------------------------
// Gather sampled centers: out[b,m,:] = xyz[b, fidx[b,m], :]
// ---------------------------------------------------------------------------
__global__ void gather_xyz_kernel(const float* __restrict__ xyz, const int* __restrict__ fidx,
                                  int N, int M, float* __restrict__ out, int total)
{
    int r = blockIdx.x * blockDim.x + threadIdx.x;
    if (r >= total) return;
    int b = r / M;
    int n = fidx[r];
    const float* s = xyz + ((size_t)b * N + n) * 3;
    out[3 * (size_t)r + 0] = s[0];
    out[3 * (size_t)r + 1] = s[1];
    out[3 * (size_t)r + 2] = s[2];
}

// ---------------------------------------------------------------------------
// Ball query: first `nsample` in-ball indices ascending, padded with the
// first in-ball index (center itself guarantees cnt >= 1).
// ---------------------------------------------------------------------------
__global__ void ball_query_kernel(const float* __restrict__ xyz, const float* __restrict__ new_xyz,
                                  int N, int M, int nsample, float r2,
                                  int* __restrict__ idx, int total)
{
    int gid = blockIdx.x * blockDim.x + threadIdx.x;
    if (gid >= total) return;
    int b = gid / M;
    const float* q = new_xyz + (size_t)gid * 3;
    float qx = q[0], qy = q[1], qz = q[2];
    const float* p = xyz + (size_t)b * N * 3;
    int* out = idx + (size_t)gid * nsample;
    int cnt = 0, first = 0;
    for (int n = 0; n < N && cnt < nsample; ++n) {
        float dx = p[3 * n + 0] - qx;
        float dy = p[3 * n + 1] - qy;
        float dz = p[3 * n + 2] - qz;
        float dd = dx * dx + dy * dy + dz * dz;
        if (dd <= r2) { if (cnt == 0) first = n; out[cnt++] = n; }
    }
    for (; cnt < nsample; ++cnt) out[cnt] = first;
}

// ---------------------------------------------------------------------------
// Grouping for SA1 (features == None): relative coords, zero-padded to 32
// channels so the downstream GEMM K-loop is guard-free (K=32).
// ---------------------------------------------------------------------------
__global__ void group_rel_xyz_kernel(const float* __restrict__ xyz, const float* __restrict__ new_xyz,
                                     const int* __restrict__ gidx, int N, int M, int ns,
                                     float* __restrict__ out, int total)
{
    int gid = blockIdx.x * blockDim.x + threadIdx.x;
    if (gid >= total) return;
    int bm = gid / ns;
    int b  = bm / M;
    int n  = gidx[gid];
    const float* p = xyz + ((size_t)b * N + n) * 3;
    const float* q = new_xyz + (size_t)bm * 3;
    float* o = out + (size_t)gid * 32;
    o[0] = p[0] - q[0];
    o[1] = p[1] - q[1];
    o[2] = p[2] - q[2];
    for (int c = 3; c < 32; ++c) o[c] = 0.0f;
}

// ---------------------------------------------------------------------------
// Fused GEMM + scale/bias + optional ReLU via v_wmma_f32_16x16x32_bf16.
//   Y[M,N] = act( (A[M,K] @ W[K,N]) * scale[n] + bias[n] )
// Preconditions: M % 64 == 0, K % 32 == 0 (A guard-free); Wp is the repacked
// bf16 fragment buffer. Block = 128 threads (4 waves x 16 cols), MT=4 M-tiles.
// EXEC is all-ones at every WMMA (guards are epilogue-only).
// ---------------------------------------------------------------------------
__global__ __launch_bounds__(128) void gemm_bn_act_wmma(
    const float* __restrict__ A, const __bf16* __restrict__ Wp,
    const float* __restrict__ scale, const float* __restrict__ bias,
    float* __restrict__ Y, int M, int N, int K, int nTiles, int doRelu)
{
    __shared__ __bf16 sA[MT * 16 * 32];

    int tid   = threadIdx.x;
    int lane  = tid & 31;
    int wave  = tid >> 5;
    int mBase = blockIdx.y * (16 * MT);
    int ntIdx = blockIdx.x * WAVES_PER_BLOCK + wave;
    int ntCl  = ntIdx < nTiles ? ntIdx : nTiles - 1;   // clamp loads; stores guarded
    int kTiles = K >> 5;

    int mloc  = lane & 15;
    int khalf = lane >> 4;

    // per-lane pointer into repacked B fragments: v16bf index (nt*kTiles+kt)*32+lane
    const v16bf* Wv = ((const v16bf*)Wp) + (size_t)ntCl * kTiles * 32 + lane;

    // A tile load pattern: thread t covers row t>>3, cols ((t&7)*4 .. +3)
    int arow = tid >> 3;
    int acol = (tid & 7) << 2;

    v8f acc[MT];
#pragma unroll
    for (int h = 0; h < MT; ++h) acc[h] = (v8f){0.f, 0.f, 0.f, 0.f, 0.f, 0.f, 0.f, 0.f};

    for (int kt = 0; kt < kTiles; ++kt) {
        int k0 = kt << 5;
        // issue all A loads first (one clause, one wait), then convert+store
        float4 av[MT];
#pragma unroll
        for (int h = 0; h < MT; ++h)
            av[h] = *(const float4*)(A + (size_t)(mBase + h * 16 + arow) * K + k0 + acol);
#pragma unroll
        for (int h = 0; h < MT; ++h) {
            v4bf pk;
            pk[0] = (__bf16)av[h].x; pk[1] = (__bf16)av[h].y;
            pk[2] = (__bf16)av[h].z; pk[3] = (__bf16)av[h].w;
            *(v4bf*)&sA[(h * 16 + arow) * 32 + acol] = pk;
        }
        __syncthreads();

        v16bf bfrag = Wv[(size_t)kt * 32];            // 2x global_load_b128
#pragma unroll
        for (int h = 0; h < MT; ++h) {
            // A-layout matching B repack: low half = K[khalf*16 .. +7], high = K[khalf*16+8 .. +7]
            v8bf alo = *(const v8bf*)&sA[(h * 16 + mloc) * 32 + khalf * 16];
            v8bf ahi = *(const v8bf*)&sA[(h * 16 + mloc) * 32 + khalf * 16 + 8];
            v16bf afrag = __builtin_shufflevector(alo, ahi, 0, 1, 2, 3, 4, 5, 6, 7,
                                                  8, 9, 10, 11, 12, 13, 14, 15);
            acc[h] = __builtin_amdgcn_wmma_f32_16x16x32_bf16(
                false, afrag, false, bfrag, (short)0, acc[h], false, false);
        }
        __syncthreads();
    }

    int n = ntIdx * 16 + mloc;
    if (ntIdx < nTiles && n < N) {
        float sc = scale ? scale[n] : 1.0f;
        float bi = bias ? bias[n] : 0.0f;
#pragma unroll
        for (int h = 0; h < MT; ++h) {
#pragma unroll
            for (int r = 0; r < 8; ++r) {
                int m = mBase + h * 16 + r + (khalf << 3);
                float y = acc[h][r] * sc + bi;
                if (doRelu) y = fmaxf(y, 0.0f);
                Y[(size_t)m * N + n] = y;
            }
        }
    }
}

// ---------------------------------------------------------------------------
// Max-pool over the nsample axis: (R, ns, C) -> (R, C)
// ---------------------------------------------------------------------------
__global__ void maxpool_ns_kernel(const float* __restrict__ in, float* __restrict__ out,
                                  int R, int ns, int C)
{
    int gid = blockIdx.x * blockDim.x + threadIdx.x;
    if (gid >= R * C) return;
    int c = gid % C, r = gid / C;
    const float* p = in + ((size_t)r * ns) * C + c;
    float mx = p[0];
    for (int s = 1; s < ns; ++s) mx = fmaxf(mx, p[(size_t)s * C]);
    out[gid] = mx;
}

// ---------------------------------------------------------------------------
// 3-NN inverse-distance interpolation (first-index ties, matching top_k).
// ---------------------------------------------------------------------------
__global__ void three_nn_interp_kernel(const float* __restrict__ xyz1, const float* __restrict__ xyz2,
                                       const float* __restrict__ f2, int N1, int N2, int C,
                                       float* __restrict__ out, int total)
{
    int gid = blockIdx.x * blockDim.x + threadIdx.x;
    if (gid >= total) return;
    int b = gid / N1;
    const float* q = xyz1 + (size_t)gid * 3;
    float qx = q[0], qy = q[1], qz = q[2];
    const float* p2 = xyz2 + (size_t)b * N2 * 3;

    float d0 = 1e30f, d1 = 1e30f, d2 = 1e30f;
    int   i0 = 0, i1 = 0, i2 = 0;
    for (int i = 0; i < N2; ++i) {
        float dx = p2[3 * i + 0] - qx;
        float dy = p2[3 * i + 1] - qy;
        float dz = p2[3 * i + 2] - qz;
        float dd = dx * dx + dy * dy + dz * dz;
        if (dd < d0)      { d2 = d1; i2 = i1; d1 = d0; i1 = i0; d0 = dd; i0 = i; }
        else if (dd < d1) { d2 = d1; i2 = i1; d1 = dd; i1 = i; }
        else if (dd < d2) { d2 = dd; i2 = i; }
    }
    float w0 = 1.0f / (fmaxf(d0, 0.0f) + 1e-8f);
    float w1 = 1.0f / (fmaxf(d1, 0.0f) + 1e-8f);
    float w2 = 1.0f / (fmaxf(d2, 0.0f) + 1e-8f);
    float ws = w0 + w1 + w2;
    w0 /= ws; w1 /= ws; w2 /= ws;

    const float* F = f2 + (size_t)b * N2 * C;
    float* o = out + (size_t)gid * C;
    for (int c = 0; c < C; ++c)
        o[c] = w0 * F[(size_t)i0 * C + c] + w1 * F[(size_t)i1 * C + c] + w2 * F[(size_t)i2 * C + c];
}

// ---------------------------------------------------------------------------
// L2 normalize rows in-place: x /= max(||x||, 1e-12)
// ---------------------------------------------------------------------------
__global__ void l2norm_kernel(float* __restrict__ x, int rows, int C)
{
    int r = blockIdx.x * blockDim.x + threadIdx.x;
    if (r >= rows) return;
    float* p = x + (size_t)r * C;
    float s = 0.0f;
    for (int c = 0; c < C; ++c) s += p[c] * p[c];
    float inv = 1.0f / fmaxf(sqrtf(s), 1e-12f);
    for (int c = 0; c < C; ++c) p[c] *= inv;
}

// ===========================================================================
// Host orchestration
// ===========================================================================
extern "C" void kernel_launch(void* const* d_in, const int* in_sizes, int n_in,
                              void* d_out, int out_size, void* d_ws, size_t ws_size,
                              hipStream_t stream)
{
    (void)in_sizes; (void)n_in; (void)out_size; (void)ws_size;

    const int B = 4, N0 = 8192, M1 = 2048, NS = 32;
    const int R1 = B * M1 * NS;   // 262144 grouped rows (SA1 GEMMs)
    const int RF = B * N0;        // 32768 dense rows (FP / heads)
    const float R2 = 0.1f * 0.1f;

    // ---- parameter leaf indexing (jax pytree: dict keys sorted) ----------
    //   d_in[0] = points; params: 'fp' < 'head' < 'sa'; layers flatten as
    //   b,beta,gamma,mean,var,w.  fp[3].L0 @ 37, fp[3].L1 @ 43;
    //   head: center{h0@49,h1@55,ob@61,ow@62} embedding{63,69,75,76}
    //         semantic{77,83,89,90};  sa[0]{91,97,103}.
    auto P = [&](int i) { return (const float*)d_in[i]; };
    const float* points = P(0);

    struct BnL { const float *b, *beta, *gamma, *mean, *var, *w; int cin, cout;
                 float *scale, *bias; __bf16* wp; };
    auto mk = [&](int base, int cin, int cout) -> BnL {
        return { P(base + 0), P(base + 1), P(base + 2), P(base + 3), P(base + 4), P(base + 5),
                 cin, cout, nullptr, nullptr, nullptr };
    };
    BnL L[11] = {
        mk(91, 3, 32),   mk(97, 32, 32),  mk(103, 32, 64),  // SA1 MLP
        mk(37, 64, 128), mk(43, 128, 128),                  // FP[3] MLP
        mk(49, 128, 256), mk(55, 256, 128),                 // center hidden
        mk(63, 128, 256), mk(69, 256, 128),                 // embedding hidden
        mk(77, 128, 256), mk(83, 256, 128),                 // semantic hidden
    };
    const float *ctr_ob = P(61), *ctr_ow = P(62);
    const float *emb_ob = P(75), *emb_ow = P(76);
    const float *sem_ob = P(89), *sem_ow = P(90);

    // ---- workspace carve-up ---------------------------------------------
    char* ws = (char*)d_ws;
    size_t off = 0;
    auto alloc = [&](size_t bytes) -> void* {
        off = (off + 255) & ~(size_t)255;
        void* p = ws + off;
        off += bytes;
        return p;
    };
    float* sbpool = (float*)alloc(2 * 4096 * sizeof(float));
    float* dist   = (float*)alloc((size_t)B * N0 * sizeof(float));
    int*   fidx   = (int*)  alloc((size_t)B * M1 * sizeof(int));
    float* xyz1   = (float*)alloc((size_t)B * M1 * 3 * sizeof(float));
    int*   gidx   = (int*)  alloc((size_t)R1 * sizeof(int));
    float* f1     = (float*)alloc((size_t)B * M1 * 64 * sizeof(float));
    float* big1   = (float*)alloc((size_t)R1 * 64 * sizeof(float)); // 67 MB
    float* big2   = (float*)alloc((size_t)R1 * 32 * sizeof(float)); // 34 MB

    float* grouped = big2;                  // (R1, 32) padded
    float* a0 = big1;                       // (R1, 32)
    float* a1 = big2;                       // (R1, 32)
    float* a2 = big1;                       // (R1, 64)
    float* X  = big2;                       // (RF, 64)
    float* g0 = big1;                       // (RF, 128)
    float* feats = big2;                    // (RF, 128)
    float* h1 = big1;                       // (RF, 256)
    float* h2 = big1 + (size_t)RF * 256;    // (RF, 128)

    // ---- fold BN + repack all weights to bf16 fragment buffers ----------
    auto repack = [&](const float* W, int K, int N) -> __bf16* {
        int kT = (K + 31) / 32, nT = (N + 15) / 16;
        size_t elems = (size_t)nT * kT * 512;
        __bf16* outp = (__bf16*)alloc(elems * sizeof(__bf16));
        repack_w_bf16_kernel<<<(int)((elems + 255) / 256), 256, 0, stream>>>(
            W, K, N, kT, nT, outp);
        return outp;
    };
    {
        size_t sb = 0;
        for (int i = 0; i < 11; ++i) {
            L[i].scale = sbpool + sb; sb += L[i].cout;
            L[i].bias  = sbpool + sb; sb += L[i].cout;
            fold_bn_kernel<<<(L[i].cout + 255) / 256, 256, 0, stream>>>(
                L[i].b, L[i].beta, L[i].gamma, L[i].mean, L[i].var,
                L[i].scale, L[i].bias, L[i].cout);
            L[i].wp = repack(L[i].w, L[i].cin, L[i].cout);
        }
    }
    __bf16* sem_wp = repack(sem_ow, 128, 13);
    __bf16* emb_wp = repack(emb_ow, 128, 32);
    __bf16* ctr_wp = repack(ctr_ow, 128, 3);

    auto gemm = [&](const float* A, const __bf16* Wp, const float* sc, const float* bi,
                    float* Y, int M, int N, int K, int relu) {
        int nT = (N + 15) / 16;
        dim3 g((nT + WAVES_PER_BLOCK - 1) / WAVES_PER_BLOCK, M / (16 * MT));
        gemm_bn_act_wmma<<<g, dim3(128), 0, stream>>>(A, Wp, sc, bi, Y, M, N, K, nT, relu);
    };

    // ---- SA module 0 -----------------------------------------------------
    fps_kernel<<<B, 1024, 0, stream>>>(points, N0, M1, fidx, dist);
    gather_xyz_kernel<<<(B * M1 + 255) / 256, 256, 0, stream>>>(points, fidx, N0, M1, xyz1, B * M1);
    ball_query_kernel<<<(B * M1 + 127) / 128, 128, 0, stream>>>(points, xyz1, N0, M1, NS, R2,
                                                                gidx, B * M1);
    group_rel_xyz_kernel<<<(R1 + 255) / 256, 256, 0, stream>>>(points, xyz1, gidx, N0, M1, NS,
                                                               grouped, R1);
    gemm(grouped, L[0].wp, L[0].scale, L[0].bias, a0, R1, 32, 32, 1); // K padded 3->32
    gemm(a0,      L[1].wp, L[1].scale, L[1].bias, a1, R1, 32, 32, 1);
    gemm(a1,      L[2].wp, L[2].scale, L[2].bias, a2, R1, 64, 32, 1);
    maxpool_ns_kernel<<<(B * M1 * 64 + 255) / 256, 256, 0, stream>>>(a2, f1, B * M1, NS, 64);

    // ---- FP module (i = 3): 3-NN interp 2048 -> 8192, then MLP [128,128] -
    three_nn_interp_kernel<<<(RF + 127) / 128, 128, 0, stream>>>(points, xyz1, f1, N0, M1, 64,
                                                                 X, RF);
    gemm(X,  L[3].wp, L[3].scale, L[3].bias, g0,    RF, 128, 64, 1);
    gemm(g0, L[4].wp, L[4].scale, L[4].bias, feats, RF, 128, 128, 1);

    // ---- heads: semantic(13) | embedding(32) | center(3) -----------------
    float* out_sem = (float*)d_out;
    float* out_emb = out_sem + (size_t)RF * 13;
    float* out_ctr = out_emb + (size_t)RF * 32;

    gemm(feats, L[9].wp,  L[9].scale,  L[9].bias,  h1, RF, 256, 128, 1);
    gemm(h1,    L[10].wp, L[10].scale, L[10].bias, h2, RF, 128, 256, 1);
    gemm(h2,    sem_wp,   nullptr,     sem_ob, out_sem, RF, 13, 128, 0);

    gemm(feats, L[7].wp,  L[7].scale,  L[7].bias,  h1, RF, 256, 128, 1);
    gemm(h1,    L[8].wp,  L[8].scale,  L[8].bias,  h2, RF, 128, 256, 1);
    gemm(h2,    emb_wp,   nullptr,     emb_ob, out_emb, RF, 32, 128, 0);

    gemm(feats, L[5].wp,  L[5].scale,  L[5].bias,  h1, RF, 256, 128, 1);
    gemm(h1,    L[6].wp,  L[6].scale,  L[6].bias,  h2, RF, 128, 256, 1);
    gemm(h2,    ctr_wp,   nullptr,     ctr_ob, out_ctr, RF, 3, 128, 0);

    l2norm_kernel<<<(RF + 127) / 128, 128, 0, stream>>>(out_emb, RF, 32);
}
